// AttentionHead_46660524703758
// MI455X (gfx1250) — matrix operand order repaired and checked by reading, not compiled, not run on hardware
//
#include <hip/hip_runtime.h>

// ---------------------------------------------------------------------------
// AttentionHead for MI455X (gfx1250, wave32, WMMA bf16 16x16x32)
// B=8, N=2048, H=1024, E=64
//
// All WMMA B-operands are pre-swizzled in memory into this code's per-lane
// register layout (k = 16*half + i): every operand fetch is 2x contiguous b128.
// Softmax runs in base-2 (log2e folded into the stored key scale).
// ---------------------------------------------------------------------------

typedef __bf16 v16bf __attribute__((ext_vector_type(16)));
typedef float  v8f   __attribute__((ext_vector_type(8)));

constexpr int Bc = 8;
constexpr int Nc = 2048;
constexpr int Hc = 1024;
constexpr int Ec = 64;

// 1/sqrt(E) * log2(e): scores land pre-scaled for exp2-based softmax
#define KSCALE 0.18033688011f

__device__ __forceinline__ unsigned short f2bf_hi(float f) {        // truncate
    return (unsigned short)(__builtin_bit_cast(unsigned, f) >> 16);
}
__device__ __forceinline__ __bf16 f2bf_trunc(float f) {
    return __builtin_bit_cast(__bf16, f2bf_hi(f));
}
// pack two floats -> two bf16 in one dword (truncation)
__device__ __forceinline__ unsigned pack_bf2(float lo, float hi) {
    return (__builtin_bit_cast(unsigned, hi) & 0xFFFF0000u) |
           (__builtin_bit_cast(unsigned, lo) >> 16);
}
__device__ __forceinline__ unsigned short f2bf_rne(float f) {       // RNE
    unsigned u = __builtin_bit_cast(unsigned, f);
    return (unsigned short)((u + 0x7FFFu + ((u >> 16) & 1u)) >> 16);
}

// contiguous 32B -> one v16bf operand (2x b128)
__device__ __forceinline__ v16bf load_v16bf(const unsigned short* p) {
    union { uint4 q[2]; v16bf v; } u;
    u.q[0] = *(const uint4*)p;
    u.q[1] = *(const uint4*)(p + 8);
    return u.v;
}

// ---------------------------------------------------------------------------
// Kernel 0: fp32 weights -> bf16 (RNE), pre-swizzled into B-operand order.
// Layout: [kc(32)][cb(4)][lane(32)][i(16)],
//   element = W[kc*32 + (lane>>4)*16 + i][cb*16 + (lane&15)]
// ---------------------------------------------------------------------------
__global__ void prep_weights(const float* __restrict__ wk,
                             const float* __restrict__ wv,
                             unsigned short* __restrict__ wkb,
                             unsigned short* __restrict__ wvb) {
    int o = blockIdx.x * blockDim.x + threadIdx.x;
    if (o >= Hc * Ec) return;
    int kc  = o >> 11;
    int rem = o & 2047;
    int cb  = rem >> 9;
    int ln  = (rem >> 4) & 31;
    int i   = o & 15;
    int kk  = kc * 32 + (ln >> 4) * 16 + i;
    int e   = cb * 16 + (ln & 15);
    int src = kk * Ec + e;
    wkb[o] = f2bf_rne(wk[src]);
    wvb[o] = f2bf_rne(wv[src]);
}

// ---------------------------------------------------------------------------
// Kernel 1: keys/values projection (16384x1024)x(1024x64), k and v fused.
// No LDS: weight B-operands come pre-swizzled from global (L2-resident).
// Epilogue:
//   keys  : row-major bf16, pre-scaled by KSCALE         (32 b16 stores)
//   valsT : B-op order for S = K * V^T                   (32 b16 stores)
//   valsB : B-op order for out += P * V, i-run contiguous ( 4 b128 stores)
// ---------------------------------------------------------------------------
__global__ void __launch_bounds__(256) project_kv(
    const float* __restrict__ inp,
    const unsigned short* __restrict__ wkb,
    const unsigned short* __restrict__ wvb,
    unsigned short* __restrict__ keys,
    unsigned short* __restrict__ valsT,
    unsigned short* __restrict__ valsB) {
  const int lane = threadIdx.x & 31;
  const int wave = threadIdx.x >> 5;
  const int half = lane >> 4;
  const int lh   = lane & 15;
  const int r0   = blockIdx.x * 128 + wave * 16;   // global row (b*N + n)

  v8f acck[4], accv[4];
  v8f zero = {};
  #pragma unroll
  for (int c = 0; c < 4; ++c) { acck[c] = zero; accv[c] = zero; }

  #pragma unroll 2
  for (int kb = 0; kb < Hc; kb += 32) {
    // A operand: fp32 -> bf16 (trunc), two contiguous float4-pairs per lane
    v16bf a;
    {
      const float* arow = inp + (size_t)(r0 + lh) * Hc + kb;
      const float4* g0 = (const float4*)(arow + half * 16);
      const float4* g1 = (const float4*)(arow + half * 16 + 8);
      float4 f0 = g0[0], f1 = g0[1], f2 = g1[0], f3 = g1[1];
      a[0]=f2bf_trunc(f0.x); a[1]=f2bf_trunc(f0.y); a[2]=f2bf_trunc(f0.z); a[3]=f2bf_trunc(f0.w);
      a[4]=f2bf_trunc(f1.x); a[5]=f2bf_trunc(f1.y); a[6]=f2bf_trunc(f1.z); a[7]=f2bf_trunc(f1.w);
      a[8]=f2bf_trunc(f2.x); a[9]=f2bf_trunc(f2.y); a[10]=f2bf_trunc(f2.z); a[11]=f2bf_trunc(f2.w);
      a[12]=f2bf_trunc(f3.x); a[13]=f2bf_trunc(f3.y); a[14]=f2bf_trunc(f3.z); a[15]=f2bf_trunc(f3.w);
    }

    const size_t wbase = (size_t)(kb >> 5) * 2048 + lane * 16;   // kc*4*512
    #pragma unroll
    for (int cb = 0; cb < 4; ++cb) {
      v16bf bk = load_v16bf(wkb + wbase + cb * 512);
      v16bf bv = load_v16bf(wvb + wbase + cb * 512);
      acck[cb] = __builtin_amdgcn_wmma_f32_16x16x32_bf16(false, a, false, bk,
                                                         (short)0, acck[cb], false, false);
      accv[cb] = __builtin_amdgcn_wmma_f32_16x16x32_bf16(false, a, false, bv,
                                                         (short)0, accv[cb], false, false);
    }
  }

  // epilogue: C/D layout -> three swizzled global layouts
  const int    b     = r0 >> 11;
  const int    n0    = r0 & (Nc - 1);
  const int    jc    = n0 >> 5;                       // constant across tile
  const size_t tb    = (size_t)(b * 64 + jc) * 2048;  // chunk base (4*512)
  const int    rbase = (n0 & 31) + half * 8;          // aligned 8-run start

  #pragma unroll
  for (int cb = 0; cb < 4; ++cb) {
    #pragma unroll
    for (int i = 0; i < 8; ++i) {
      const int    rr = rbase + i;                    // row within 32-chunk
      const size_t m  = (size_t)r0 + i + half * 8;    // global row
      const int    e  = cb * 16 + lh;
      // keys row-major, pre-scaled (exp2-softmax scale folded in)
      keys[m * Ec + e] = f2bf_hi(acck[cb][i] * KSCALE);
      // valsT: [cbT=rr>>4][ecT=cb>>1][laneT=(cb&1)*16+(rr&15)][iT=lh]
      valsT[tb + (size_t)(((rr >> 4) * 2 + (cb >> 1)) * 512)
               + ((cb & 1) * 16 + (rr & 15)) * 16 + lh] = f2bf_hi(accv[cb][i]);
    }
    // valsB: [eb=cb][laneB=(rbase>>4)*16+lh][iB=rbase&15 .. +7] contiguous
    uint4 q;
    q.x = pack_bf2(accv[cb][0], accv[cb][1]);
    q.y = pack_bf2(accv[cb][2], accv[cb][3]);
    q.z = pack_bf2(accv[cb][4], accv[cb][5]);
    q.w = pack_bf2(accv[cb][6], accv[cb][7]);
    *(uint4*)(valsB + tb + (size_t)(cb * 512)
                    + ((rbase >> 4) * 16 + lh) * 16 + (rbase & 15)) = q;
  }
}

// ---------------------------------------------------------------------------
// Kernel 2: flash attention. 4 independent waves per block, one 16-row key
// tile each. S = (K*KSCALE) @ V^T (base-2 domain), mask keeps col >= row,
// online softmax via exp2, out = attn @ values. All V operands are direct
// b128 global loads; PV operands prefetched into registers before softmax.
// Only LDS use: per-wave 16x32 P tile re-swizzle into A-operand layout.
// ---------------------------------------------------------------------------
__global__ void __launch_bounds__(128) attention(
    const unsigned short* __restrict__ keys,
    const unsigned short* __restrict__ valsT,
    const unsigned short* __restrict__ valsB,
    float* __restrict__ out) {
  __shared__ __align__(16) float Ps[4 * 16 * 32];

  const int lane = threadIdx.x & 31;
  const int wave = threadIdx.x >> 5;
  const int half = lane >> 4;
  const int lh   = lane & 15;
  const int b    = blockIdx.y;
  const int i0   = blockIdx.x * 64 + wave * 16;
  float* myPs    = Ps + wave * 512;
  const size_t rowbase = (size_t)b * Nc;

  // keys tile as two A operands (e-chunks of 32), contiguous b128 loads
  v16bf Ak[2];
  {
    const unsigned short* krow = keys + (rowbase + i0 + lh) * Ec;
    #pragma unroll
    for (int ec = 0; ec < 2; ++ec) {
      union { uint4 q[2]; v16bf v; } u;
      u.q[0] = *(const uint4*)(krow + ec * 32 + half * 16);
      u.q[1] = *(const uint4*)(krow + ec * 32 + half * 16 + 8);
      Ak[ec] = u.v;
    }
  }

  float mrow[8], lrow[8];
  v8f acco[4];
  v8f zero = {};
  #pragma unroll
  for (int i = 0; i < 8; ++i) { mrow[i] = -3.0e38f; lrow[i] = 0.0f; }
  #pragma unroll
  for (int c = 0; c < 4; ++c) acco[c] = zero;

  // mask keeps col >= row: columns < (i0 & ~31) are fully masked -> skipped
  for (int j0 = i0 & ~31; j0 < Nc; j0 += 32) {
    const size_t cbase = (size_t)(b * 64 + (j0 >> 5)) * 2048 + lane * 16;

    // issue ALL operand loads up front: bt needed now, bv after softmax
    v16bf bt[4], bv[4];
    #pragma unroll
    for (int t = 0; t < 4; ++t) bt[t] = load_v16bf(valsT + cbase + t * 512);
    #pragma unroll
    for (int t = 0; t < 4; ++t) bv[t] = load_v16bf(valsB + cbase + t * 512);

    // prefetch next j-block's operand chunks (4KB each, 128B per lane)
    if (j0 + 32 < Nc) {
      const size_t nb = cbase - lane * 16 + 2048 + lane * 64;
      __builtin_prefetch(valsT + nb, 0, 1);
      __builtin_prefetch(valsB + nb, 0, 1);
    }

    // S[cb] = K_scaled @ V^T   (already in log2 domain)
    v8f S[2];
    #pragma unroll
    for (int cb = 0; cb < 2; ++cb) {
      v8f s = zero;
      #pragma unroll
      for (int ec = 0; ec < 2; ++ec)
        s = __builtin_amdgcn_wmma_f32_16x16x32_bf16(false, Ak[ec], false,
                                                    bt[cb * 2 + ec],
                                                    (short)0, s, false, false);
      S[cb] = s;
    }

    if (j0 < i0 + 16) {   // only diagonal-overlapping blocks need the mask
      #pragma unroll
      for (int cb = 0; cb < 2; ++cb) {
        const int c = j0 + cb * 16 + lh;
        #pragma unroll
        for (int i = 0; i < 8; ++i) {
          const int r = i0 + i + half * 8;
          if (c < r) S[cb][i] = -3.0e38f;
        }
      }
    }

    // online softmax (base-2); reductions across 16 lanes of each half
    #pragma unroll
    for (int i = 0; i < 8; ++i) {
      float mx = fmaxf(S[0][i], S[1][i]);
      #pragma unroll
      for (int m = 1; m < 16; m <<= 1) mx = fmaxf(mx, __shfl_xor(mx, m, 32));
      const float mnew  = fmaxf(mrow[i], mx);
      const float alpha = __builtin_amdgcn_exp2f(mrow[i] - mnew);
      const float p0 = __builtin_amdgcn_exp2f(S[0][i] - mnew);
      const float p1 = __builtin_amdgcn_exp2f(S[1][i] - mnew);
      float rs = p0 + p1;
      #pragma unroll
      for (int m = 1; m < 16; m <<= 1) rs += __shfl_xor(rs, m, 32);
      lrow[i] = fmaf(lrow[i], alpha, rs);
      mrow[i] = mnew;
      #pragma unroll
      for (int c = 0; c < 4; ++c) acco[c][i] *= alpha;
      myPs[(i + half * 8) * 32 + lh]      = p0;   // C/D layout -> row-major
      myPs[(i + half * 8) * 32 + 16 + lh] = p1;
    }
    // per-wave LDS visibility (no cross-wave sharing -> no barrier needed)
    asm volatile("s_wait_dscnt 0" ::: "memory");

    // P -> A-operand layout (contiguous f32 groups -> b128 LDS loads)
    v16bf Ap;
    {
      const float* p0 = myPs + lh * 32 + half * 16;
      const float* p1 = myPs + lh * 32 + half * 16 + 8;
      #pragma unroll
      for (int i = 0; i < 8; ++i) {
        Ap[i]     = f2bf_trunc(p0[i]);
        Ap[8 + i] = f2bf_trunc(p1[i]);
      }
    }
    asm volatile("" ::: "memory");   // keep next iter's Ps stores below reads

    // out(16x64) += P(16x32) @ V(32x64) : operands already in registers
    #pragma unroll
    for (int eb = 0; eb < 4; ++eb)
      acco[eb] = __builtin_amdgcn_wmma_f32_16x16x32_bf16(false, Ap, false, bv[eb],
                                                         (short)0, acco[eb], false, false);
  }

  #pragma unroll
  for (int i = 0; i < 8; ++i) {
    const float inv = __builtin_amdgcn_rcpf(lrow[i]);
    const size_t row = rowbase + i0 + i + half * 8;
    #pragma unroll
    for (int eb = 0; eb < 4; ++eb)
      out[row * Ec + eb * 16 + lh] = acco[eb][i] * inv;
  }
}

// ---------------------------------------------------------------------------
extern "C" void kernel_launch(void* const* d_in, const int* in_sizes, int n_in,
                              void* d_out, int out_size, void* d_ws, size_t ws_size,
                              hipStream_t stream) {
  (void)in_sizes; (void)n_in; (void)out_size; (void)ws_size;

  const float* inp = (const float*)d_in[0];
  const float* wk  = (const float*)d_in[1];
  // d_in[2] = q : reference computes `queries` but never uses it -> skipped
  const float* wv  = (const float*)d_in[3];
  float* out = (float*)d_out;

  // workspace (bf16): wk_swz | wv_swz | keys | valsT | valsB  (~6.5 MB)
  unsigned short* ws    = (unsigned short*)d_ws;
  unsigned short* wkb   = ws;
  unsigned short* wvb   = wkb + (size_t)Hc * Ec;
  unsigned short* keys  = wvb + (size_t)Hc * Ec;
  unsigned short* valsT = keys + (size_t)Bc * Nc * Ec;
  unsigned short* valsB = valsT + (size_t)Bc * Nc * Ec;

  prep_weights<<<(Hc * Ec + 255) / 256, 256, 0, stream>>>(wk, wv, wkb, wvb);
  project_kv<<<(Bc * Nc) / 128, 256, 0, stream>>>(inp, wkb, wvb, keys, valsT, valsB);
  dim3 grid(Nc / 64, Bc);
  attention<<<grid, 128, 0, stream>>>(keys, valsT, valsB, out);
}
